// T5Model_21500606284394
// MI455X (gfx1250) — matrix operand order repaired and checked by reading, not compiled, not run on hardware
//
#include <hip/hip_runtime.h>

// ---------------------------------------------------------------------------
// T5 (B=4, S=T=512, H=512, F=1024, NH=8, dh=64, L=6+6, V=32000) forward
// for gfx1250: all GEMMs through V_WMMA_F32_16X16X32_F16 (f32 accum),
// f32->f16 conversion fused into a fully vectorized, double-buffered LDS
// staging pipeline (global_load_b128 -> regs -> ds_store_b64).
//
// Shape contract used by every launch in this file (lets us drop guards):
//   M % 128 == 0, K % 32 == 0, and either N % 128 == 0 or (non-transB with
//   ldb wide enough that OOB columns are still in-bounds reads; AV: N=64,
//   ldb=512).  Only C stores are predicated on n < N.
// ---------------------------------------------------------------------------

typedef __attribute__((ext_vector_type(16))) _Float16 v16h;
typedef __attribute__((ext_vector_type(8)))  _Float16 v8h;
typedef __attribute__((ext_vector_type(4)))  _Float16 v4h;
typedef __attribute__((ext_vector_type(8)))  float    v8f;

enum GemmFlags {
  F_TRANSB  = 1,   // B stored [N,K] instead of [K,N]
  F_RELU    = 2,
  F_BIAS    = 4,   // += bias[n]
  F_REL     = 8,   // += rel[bucket[m*N+n]*8 + headIdx]   (T5 relative bias)
  F_MASKENC = 16,  // += (1 - mask[b*N+n]) * -1e4          (padding mask)
  F_CAUSAL  = 32,  // += (n > m) ? -1e4 : 0
};

#define TILE_M 128
#define TILE_N 128
#define TILE_K 32
#define LDSL   40   // halfs per LDS row: 32 data + 8 pad (80B, 16B-aligned)

// ---------------------------------------------------------------------------
// Batched GEMM: C[M,N] = A[M,K] * B (+ fused epilogue). 8 waves, 128x128 tile,
// each wave computes a 32x64 sub-tile = 2x4 WMMA accumulators. Batch index z
// decomposes as (outer = z/innerB, inner = z%innerB) with separate strides so
// per-(batch,head) slices of [B,S,H] tensors address directly.
// ---------------------------------------------------------------------------
__global__ __launch_bounds__(256)
void gemm_wmma_kernel(const float* __restrict__ A, const float* __restrict__ Bm,
                      float* __restrict__ C,
                      int M, int N, int K, int lda, int ldb, int ldc,
                      int innerB,
                      long long sAo, long long sAi,
                      long long sBo, long long sBi,
                      long long sCo, long long sCi,
                      int flags,
                      const float* __restrict__ bias,
                      const float* __restrict__ rel,
                      const int*   __restrict__ bucket,
                      const float* __restrict__ mask)
{
  __shared__ _Float16 As[2][TILE_M * LDSL];   // As[buf][m][k]
  __shared__ _Float16 Bs[2][TILE_N * LDSL];   // Bs[buf][n][k] (B transposed)

  const int tid   = threadIdx.x;
  const int lane  = tid & 31;
  const int wave  = tid >> 5;
  const int wm    = (wave & 3) * 32;   // wave row offset inside tile
  const int wn    = (wave >> 2) * 64;  // wave col offset inside tile
  const int tileM = blockIdx.y * TILE_M;
  const int tileN = blockIdx.x * TILE_N;

  const int z  = (int)blockIdx.z;
  const int bO = z / innerB;
  const int hI = z - bO * innerB;
  const float* Ab = A  + bO * sAo + hI * sAi;
  const float* Bb = Bm + bO * sBo + hI * sBi;
  float*       Cb = C  + bO * sCo + hI * sCi;
  const bool transB = (flags & F_TRANSB) != 0;

  // ---- per-thread staging coordinates (hoisted) ---------------------------
  // A tile: 128 rows x 8 float4/row; thread handles 4 float4s.
  const int a_r  = tid >> 3;           // base row for i=0 (rows step +32)
  const int a_c4 = tid & 7;            // float4 column 0..7
  // B transposed-staging (B stored [N,K]): same geometry, rows are n.
  // B normal ([K,N]): 32 k-rows x 32 float4/row.
  const int b_kr  = tid >> 5;          // k row for i=0 (rows step +8)
  const int b_nc4 = tid & 31;          // float4 col 0..31 -> n = nc4*4

  float4 aR[4], bR[4];

  // ---- staging helpers ----------------------------------------------------
  auto loadTiles = [&](int k0) {
#pragma unroll
    for (int i = 0; i < 4; ++i) {
      int r = a_r + i * 32;
      aR[i] = *(const float4*)(Ab + (long long)(tileM + r) * lda + (k0 + a_c4 * 4));
    }
    if (transB) {
#pragma unroll
      for (int i = 0; i < 4; ++i) {
        int r = a_r + i * 32;          // r = n inside tile (N % 128 == 0 here)
        bR[i] = *(const float4*)(Bb + (long long)(tileN + r) * ldb + (k0 + a_c4 * 4));
      }
    } else {
      int gn4 = tileN + b_nc4 * 4;
      float msk = (gn4 < N) ? 1.0f : 0.0f;   // branch-free OOB zeroing
#pragma unroll
      for (int i = 0; i < 4; ++i) {
        int kr = b_kr + i * 8;
        float4 t = *(const float4*)(Bb + (long long)(k0 + kr) * ldb + gn4);
        t.x *= msk; t.y *= msk; t.z *= msk; t.w *= msk;
        bR[i] = t;
      }
    }
  };

  auto storeTiles = [&](int buf) {
#pragma unroll
    for (int i = 0; i < 4; ++i) {
      int r = a_r + i * 32;
      v4h h; h[0] = (_Float16)aR[i].x; h[1] = (_Float16)aR[i].y;
             h[2] = (_Float16)aR[i].z; h[3] = (_Float16)aR[i].w;
      *(v4h*)&As[buf][r * LDSL + a_c4 * 4] = h;       // ds_store_b64
    }
    if (transB) {
#pragma unroll
      for (int i = 0; i < 4; ++i) {
        int r = a_r + i * 32;
        v4h h; h[0] = (_Float16)bR[i].x; h[1] = (_Float16)bR[i].y;
               h[2] = (_Float16)bR[i].z; h[3] = (_Float16)bR[i].w;
        *(v4h*)&Bs[buf][r * LDSL + a_c4 * 4] = h;     // ds_store_b64
      }
    } else {
#pragma unroll
      for (int i = 0; i < 4; ++i) {                   // transpose scatter
        int kr = b_kr + i * 8;
        int n0 = b_nc4 * 4;
        Bs[buf][(n0 + 0) * LDSL + kr] = (_Float16)bR[i].x;
        Bs[buf][(n0 + 1) * LDSL + kr] = (_Float16)bR[i].y;
        Bs[buf][(n0 + 2) * LDSL + kr] = (_Float16)bR[i].z;
        Bs[buf][(n0 + 3) * LDSL + kr] = (_Float16)bR[i].w;
      }
    }
  };

  // ---- accumulators -------------------------------------------------------
  const v8f vzero = {0.f,0.f,0.f,0.f,0.f,0.f,0.f,0.f};
  v8f acc[2][4];
#pragma unroll
  for (int i = 0; i < 2; ++i)
#pragma unroll
    for (int j = 0; j < 4; ++j) acc[i][j] = vzero;

  const int m16 = lane & 15;   // row (A) / col (B) inside 16-wide fragment
  const int sel = lane >> 4;   // K-half select per ISA 16-bit operand layout

  // ---- double-buffered main loop -----------------------------------------
  loadTiles(0);
  storeTiles(0);
  __syncthreads();

  int cur = 0;
  for (int k0 = 0;;) {
    const int k1 = k0 + TILE_K;
    const bool more = k1 < K;
    if (more) loadTiles(k1);           // HBM latency hides behind WMMAs below

    const _Float16* Ac = &As[cur][0];
    const _Float16* Bc = &Bs[cur][0];
    v16h aF[2], bF[4];
#pragma unroll
    for (int i = 0; i < 2; ++i) {
      const _Float16* p = &Ac[(wm + i * 16 + m16) * LDSL + sel * 8];
      v8h lo = *(const v8h*)p;          // K = sel*8 + 0..7
      v8h hi = *(const v8h*)(p + 16);   // K = 16 + sel*8 + 0..7
#pragma unroll
      for (int t = 0; t < 8; ++t) { aF[i][t] = lo[t]; aF[i][t + 8] = hi[t]; }
    }
#pragma unroll
    for (int j = 0; j < 4; ++j) {
      const _Float16* p = &Bc[(wn + j * 16 + m16) * LDSL + sel * 8];
      v8h lo = *(const v8h*)p;          // K = sel*8 + 0..7
      v8h hi = *(const v8h*)(p + 16);   // K = 16 + sel*8 + 0..7
#pragma unroll
      for (int t = 0; t < 8; ++t) { bF[j][t] = lo[t]; bF[j][t + 8] = hi[t]; }
    }
#pragma unroll
    for (int i = 0; i < 2; ++i)
#pragma unroll
      for (int j = 0; j < 4; ++j)
        acc[i][j] = __builtin_amdgcn_wmma_f32_16x16x32_f16(
            false, aF[i], false, bF[j], (short)0, acc[i][j], false, false);

    if (!more) break;
    __syncthreads();                    // everyone done reading As/Bs[cur^1]
    storeTiles(cur ^ 1);
    __syncthreads();                    // tile (k1) visible
    cur ^= 1;
    k0 = k1;
  }

  // ---- epilogue: C/D layout n = lane&15, M = r + 8*(lane>>4) --------------
#pragma unroll
  for (int i = 0; i < 2; ++i) {
#pragma unroll
    for (int j = 0; j < 4; ++j) {
      int nn = tileN + wn + j * 16 + (lane & 15);
#pragma unroll
      for (int r = 0; r < 8; ++r) {
        int mm = tileM + wm + i * 16 + r + 8 * (lane >> 4);
        if (nn < N) {                   // M % 128 == 0: no m guard needed
          float v = acc[i][j][r];
          if (flags & F_BIAS)    v += bias[nn];
          if (flags & F_REL)     v += rel[bucket[mm * N + nn] * 8 + hI];
          if (flags & F_MASKENC) v += (1.0f - mask[bO * N + nn]) * -10000.0f;
          if ((flags & F_CAUSAL) && (nn > mm)) v += -10000.0f;
          if (flags & F_RELU)    v = fmaxf(v, 0.0f);
          Cb[(long long)mm * ldc + nn] = v;
        }
      }
    }
  }
}

// ---------------------------------------------------------------------------
// Wave-per-row softmax over 512 columns (16 elems/lane, shfl_xor reductions)
// ---------------------------------------------------------------------------
__global__ __launch_bounds__(256)
void softmax512_kernel(float* __restrict__ x, int rows)
{
  int row  = blockIdx.x * 8 + (threadIdx.x >> 5);
  int lane = threadIdx.x & 31;
  if (row >= rows) return;
  float* p = x + (long long)row * 512;
  float v[16];
  float mx = -3.0e38f;
#pragma unroll
  for (int i = 0; i < 16; ++i) { v[i] = p[lane + i * 32]; mx = fmaxf(mx, v[i]); }
#pragma unroll
  for (int o = 16; o >= 1; o >>= 1) mx = fmaxf(mx, __shfl_xor(mx, o, 32));
  float s = 0.0f;
#pragma unroll
  for (int i = 0; i < 16; ++i) { v[i] = __expf(v[i] - mx); s += v[i]; }
#pragma unroll
  for (int o = 16; o >= 1; o >>= 1) s += __shfl_xor(s, o, 32);
  float inv = 1.0f / s;
#pragma unroll
  for (int i = 0; i < 16; ++i) p[lane + i * 32] = v[i] * inv;
}

// ---------------------------------------------------------------------------
// Wave-per-row residual-add + LayerNorm over H=512 (res may be null)
// ---------------------------------------------------------------------------
__global__ __launch_bounds__(256)
void add_ln_kernel(const float* __restrict__ x, const float* __restrict__ res,
                   const float* __restrict__ g, const float* __restrict__ b,
                   float* __restrict__ out, int rows)
{
  int row  = blockIdx.x * 8 + (threadIdx.x >> 5);
  int lane = threadIdx.x & 31;
  if (row >= rows) return;
  const float* px = x + (long long)row * 512;
  const float* pr = res ? res + (long long)row * 512 : nullptr;
  float v[16];
  float s = 0.0f;
#pragma unroll
  for (int i = 0; i < 16; ++i) {
    float t = px[lane + i * 32];
    if (pr) t += pr[lane + i * 32];
    v[i] = t; s += t;
  }
#pragma unroll
  for (int o = 16; o >= 1; o >>= 1) s += __shfl_xor(s, o, 32);
  float mu = s * (1.0f / 512.0f);
  float var = 0.0f;
#pragma unroll
  for (int i = 0; i < 16; ++i) { float d = v[i] - mu; var += d * d; }
#pragma unroll
  for (int o = 16; o >= 1; o >>= 1) var += __shfl_xor(var, o, 32);
  var *= (1.0f / 512.0f);
  float rinv = rsqrtf(var + 1e-5f);
  float* po = out + (long long)row * 512;
#pragma unroll
  for (int i = 0; i < 16; ++i) {
    int col = lane + i * 32;
    po[col] = (v[i] - mu) * rinv * g[col] + b[col];
  }
}

// ---------------------------------------------------------------------------
// Embedding gather: out[row,:] = emb[ids[row],:]  (H=512, float4 copies)
// ---------------------------------------------------------------------------
__global__ __launch_bounds__(128)
void embed_kernel(const int* __restrict__ ids, const float* __restrict__ emb,
                  float* __restrict__ out)
{
  int row = blockIdx.x;
  int t   = threadIdx.x;            // 128 threads * 4 floats = 512
  long long id = ids[row];
  const float4* src = (const float4*)(emb + id * 512);
  float4*       dst = (float4*)(out + (long long)row * 512);
  dst[t] = src[t];
}

// ---------------------------------------------------------------------------
// T5 bidirectional relative-position bucket table, S=512 (shared enc/dec size)
// ---------------------------------------------------------------------------
__global__ __launch_bounds__(256)
void bucket_kernel(int* __restrict__ bucket)
{
  int idx = blockIdx.x * 256 + threadIdx.x;
  if (idx >= 512 * 512) return;
  int q = idx >> 9, k = idx & 511;
  int n = q - k;                    // n = -(mem - ctx)
  int ret = (n < 0) ? 16 : 0;
  n = (n < 0) ? -n : n;
  int bb;
  if (n < 8) {
    bb = n;
  } else {
    bb = 8 + (int)(__logf((float)n * 0.125f) * (8.0f / __logf(16.0f)));
    if (bb > 15) bb = 15;
  }
  bucket[idx] = ret + bb;
}

// ---------------------------------------------------------------------------
// Host-side helpers (everything on `stream`; graph-capture safe)
// ---------------------------------------------------------------------------
static inline void gemm(hipStream_t st, const float* A, const float* B, float* C,
                        int M, int N, int K, int lda, int ldb, int ldc,
                        int batch, int innerB,
                        long long sAo, long long sAi,
                        long long sBo, long long sBi,
                        long long sCo, long long sCi,
                        int flags,
                        const float* bias, const float* rel,
                        const int* bucket, const float* mask)
{
  dim3 grid((N + TILE_N - 1) / TILE_N, (M + TILE_M - 1) / TILE_M, batch);
  gemm_wmma_kernel<<<grid, 256, 0, st>>>(A, B, C, M, N, K, lda, ldb, ldc,
                                         innerB, sAo, sAi, sBo, sBi, sCo, sCi,
                                         flags, bias, rel, bucket, mask);
}

static void run_attention(hipStream_t st, const float* xq, const float* xkv,
                          const float* w4 /* [4,H,H] */,
                          const float* rel /* [32,8] or null */,
                          const int* bucket, const float* mask /* or null */,
                          bool causal,
                          float* q, float* k, float* v,
                          float* scores, float* ctx, float* aout)
{
  const int Bsz = 4, S = 512, H = 512, NH = 8, dh = 64;
  const long long HH = (long long)H * H;
  // projections
  gemm(st, xq,  w4 + 0 * HH, q, Bsz * S, H, H, H, H, H, 1, 1,
       0, 0, 0, 0, 0, 0, 0, nullptr, nullptr, nullptr, nullptr);
  gemm(st, xkv, w4 + 1 * HH, k, Bsz * S, H, H, H, H, H, 1, 1,
       0, 0, 0, 0, 0, 0, 0, nullptr, nullptr, nullptr, nullptr);
  gemm(st, xkv, w4 + 2 * HH, v, Bsz * S, H, H, H, H, H, 1, 1,
       0, 0, 0, 0, 0, 0, 0, nullptr, nullptr, nullptr, nullptr);
  // scores[b,h] = q_slice [S,dh] * k_slice^T  (+ rel bias + masks fused)
  int sflags = F_TRANSB | (rel ? F_REL : 0) | (mask ? F_MASKENC : 0)
             | (causal ? F_CAUSAL : 0);
  gemm(st, q, k, scores, S, S, dh, H, H, S, Bsz * NH, NH,
       (long long)S * H, dh, (long long)S * H, dh,
       (long long)NH * S * S, (long long)S * S,
       sflags, nullptr, rel, bucket, mask);
  softmax512_kernel<<<(Bsz * NH * S) / 8, 256, 0, st>>>(scores, Bsz * NH * S);
  // ctx[b,h] = aw [S,S] * v_slice [S,dh], written into [B,S,H] head columns
  gemm(st, scores, v, ctx, S, dh, S, S, H, H, Bsz * NH, NH,
       (long long)NH * S * S, (long long)S * S,
       (long long)S * H, dh,
       (long long)S * H, dh,
       0, nullptr, nullptr, nullptr, nullptr);
  // output projection
  gemm(st, ctx, w4 + 3 * HH, aout, Bsz * S, H, H, H, H, H, 1, 1,
       0, 0, 0, 0, 0, 0, 0, nullptr, nullptr, nullptr, nullptr);
}

extern "C" void kernel_launch(void* const* d_in, const int* in_sizes, int n_in,
                              void* d_out, int out_size, void* d_ws, size_t ws_size,
                              hipStream_t stream)
{
  (void)in_sizes; (void)n_in; (void)out_size; (void)ws_size;
  const int Bsz = 4, S = 512, H = 512, F = 1024, L = 6, NH = 8, NB = 32;

  const int*   input_ids   = (const int*)  d_in[0];
  const int*   dec_ids     = (const int*)  d_in[1];
  const float* enc_mask    = (const float*)d_in[2];
  const float* emb         = (const float*)d_in[3];
  const float* enc_attn_w  = (const float*)d_in[4];
  const float* enc_rel     = (const float*)d_in[5];
  const float* enc_ln_g    = (const float*)d_in[6];
  const float* enc_ln_b    = (const float*)d_in[7];
  const float* enc_ffn_w1  = (const float*)d_in[8];
  const float* enc_ffn_b1  = (const float*)d_in[9];
  const float* enc_ffn_w2  = (const float*)d_in[10];
  const float* enc_ffn_b2  = (const float*)d_in[11];
  const float* enc_final_g = (const float*)d_in[12];
  const float* enc_final_b = (const float*)d_in[13];
  const float* dec_self_w  = (const float*)d_in[14];
  const float* dec_rel     = (const float*)d_in[15];
  const float* dec_cross_w = (const float*)d_in[16];
  const float* dec_ln_g    = (const float*)d_in[17];
  const float* dec_ln_b    = (const float*)d_in[18];
  const float* dec_ffn_w1  = (const float*)d_in[19];
  const float* dec_ffn_b1  = (const float*)d_in[20];
  const float* dec_ffn_w2  = (const float*)d_in[21];
  const float* dec_ffn_b2  = (const float*)d_in[22];
  const float* dec_final_g = (const float*)d_in[23];
  const float* dec_final_b = (const float*)d_in[24];

  // ---- workspace carve (~76 MB) ------------------------------------------
  float* ws = (float*)d_ws;
  size_t off = 0;
  const size_t NTOK = (size_t)Bsz * S;       // 2048 rows
  float* hEnc   = ws + off; off += NTOK * H;
  float* encOut = ws + off; off += NTOK * H;
  float* hDec   = ws + off; off += NTOK * H;
  float* qb     = ws + off; off += NTOK * H;
  float* kb     = ws + off; off += NTOK * H;
  float* vb     = ws + off; off += NTOK * H;
  float* ctx    = ws + off; off += NTOK * H;
  float* proj   = ws + off; off += NTOK * H;
  float* ffn    = ws + off; off += NTOK * F;
  float* scores = ws + off; off += (size_t)Bsz * NH * S * S;
  int*   bucket = (int*)(ws + off); off += (size_t)S * S;

  const long long HH = (long long)H * H;
  const int LN_BLKS = (int)(NTOK / 8);

  bucket_kernel<<<(S * S) / 256, 256, 0, stream>>>(bucket);
  embed_kernel<<<(int)NTOK, 128, 0, stream>>>(input_ids, emb, hEnc);
  embed_kernel<<<(int)NTOK, 128, 0, stream>>>(dec_ids, emb, hDec);

  // ---- encoder ------------------------------------------------------------
  for (int l = 0; l < L; ++l) {
    run_attention(stream, hEnc, hEnc, enc_attn_w + (long long)l * 4 * HH,
                  enc_rel + (size_t)l * NB * NH, bucket, enc_mask, false,
                  qb, kb, vb, scores, ctx, proj);
    add_ln_kernel<<<LN_BLKS, 256, 0, stream>>>(
        hEnc, proj, enc_ln_g + (size_t)l * 2 * H, enc_ln_b + (size_t)l * 2 * H,
        hEnc, (int)NTOK);
    gemm(stream, hEnc, enc_ffn_w1 + (long long)l * H * F, ffn,
         (int)NTOK, F, H, H, F, F, 1, 1, 0, 0, 0, 0, 0, 0,
         F_BIAS | F_RELU, enc_ffn_b1 + (size_t)l * F, nullptr, nullptr, nullptr);
    gemm(stream, ffn, enc_ffn_w2 + (long long)l * F * H, proj,
         (int)NTOK, H, F, F, H, H, 1, 1, 0, 0, 0, 0, 0, 0,
         F_BIAS, enc_ffn_b2 + (size_t)l * H, nullptr, nullptr, nullptr);
    add_ln_kernel<<<LN_BLKS, 256, 0, stream>>>(
        hEnc, proj, enc_ln_g + (size_t)l * 2 * H + H,
        enc_ln_b + (size_t)l * 2 * H + H, hEnc, (int)NTOK);
  }
  add_ln_kernel<<<LN_BLKS, 256, 0, stream>>>(
      hEnc, nullptr, enc_final_g, enc_final_b, encOut, (int)NTOK);

  // ---- decoder ------------------------------------------------------------
  for (int l = 0; l < L; ++l) {
    // causal self-attention with relative bias
    run_attention(stream, hDec, hDec, dec_self_w + (long long)l * 4 * HH,
                  dec_rel + (size_t)l * NB * NH, bucket, nullptr, true,
                  qb, kb, vb, scores, ctx, proj);
    add_ln_kernel<<<LN_BLKS, 256, 0, stream>>>(
        hDec, proj, dec_ln_g + (size_t)l * 3 * H, dec_ln_b + (size_t)l * 3 * H,
        hDec, (int)NTOK);
    // cross-attention over encoder output (padding mask, no rel bias)
    run_attention(stream, hDec, encOut, dec_cross_w + (long long)l * 4 * HH,
                  nullptr, bucket, enc_mask, false,
                  qb, kb, vb, scores, ctx, proj);
    add_ln_kernel<<<LN_BLKS, 256, 0, stream>>>(
        hDec, proj, dec_ln_g + (size_t)l * 3 * H + H,
        dec_ln_b + (size_t)l * 3 * H + H, hDec, (int)NTOK);
    // FFN
    gemm(stream, hDec, dec_ffn_w1 + (long long)l * H * F, ffn,
         (int)NTOK, F, H, H, F, F, 1, 1, 0, 0, 0, 0, 0, 0,
         F_BIAS | F_RELU, dec_ffn_b1 + (size_t)l * F, nullptr, nullptr, nullptr);
    gemm(stream, ffn, dec_ffn_w2 + (long long)l * F * H, proj,
         (int)NTOK, H, F, F, H, H, 1, 1, 0, 0, 0, 0, 0, 0,
         F_BIAS, dec_ffn_b2 + (size_t)l * H, nullptr, nullptr, nullptr);
    add_ln_kernel<<<LN_BLKS, 256, 0, stream>>>(
        hDec, proj, dec_ln_g + (size_t)l * 3 * H + 2 * H,
        dec_ln_b + (size_t)l * 3 * H + 2 * H, hDec, (int)NTOK);
  }
  add_ln_kernel<<<LN_BLKS, 256, 0, stream>>>(
      hDec, nullptr, dec_final_g, dec_final_b, hDec, (int)NTOK);

  // ---- tied LM head: logits = d @ emb^T  ([2048,512] x [512,32000]) -------
  gemm(stream, hDec, emb, (float*)d_out, (int)NTOK, 32000, H, H, H, 32000,
       1, 1, 0, 0, 0, 0, 0, 0, F_TRANSB, nullptr, nullptr, nullptr, nullptr);
}
